// EncoderLayer_1657857376763
// MI455X (gfx1250) — matrix operand order repaired and checked by reading, not compiled, not run on hardware
//
#include <hip/hip_runtime.h>
#include <hip/hip_bf16.h>
#include <math.h>

// ---------------------------------------------------------------------------
// Transformer encoder layer on MI455X (gfx1250, wave32, WMMA bf16 16x16x32).
// ~120 GFLOP, all operands L2-resident (192MB) -> compute-bound -> WMMA.
// LDS-staged attention tiles, double-buffered GEMM with CDNA5 async-to-LDS
// copies (global_load_async_to_lds_b128 + s_wait_asynccnt), L2 prefetch.
// ---------------------------------------------------------------------------

typedef __bf16 bf16_t;
typedef __attribute__((ext_vector_type(16))) __bf16 v16bf;
typedef __attribute__((ext_vector_type(8)))  float  v8f;
typedef __attribute__((ext_vector_type(4)))  int    v4i;

#define D_MODEL 1024
#define N_HEADS 16
#define HDIM    64
#define HIDDEN  4096
#define SEQ     1024
#define BATCH   4
#define NTOK    (BATCH * SEQ)
#define LN_EPS  1e-12f

#define AS1 __attribute__((address_space(1)))
#define AS3 __attribute__((address_space(3)))

#if __has_builtin(__builtin_amdgcn_global_load_async_to_lds_b128)
#define HAVE_ASYNC_COPY 1
#else
#define HAVE_ASYNC_COPY 0
#endif

#if __has_builtin(__builtin_amdgcn_s_wait_asynccnt)
#define WAIT_ASYNC() __builtin_amdgcn_s_wait_asynccnt(0)
#elif HAVE_ASYNC_COPY
#define WAIT_ASYNC() asm volatile("s_wait_asynccnt 0" ::: "memory")
#else
#define WAIT_ASYNC()
#endif

// ---------------------------------------------------------------------------
// WMMA helpers
// ---------------------------------------------------------------------------
__device__ __forceinline__ v8f zero_v8f() {
  v8f z;
#pragma unroll
  for (int i = 0; i < 8; ++i) z[i] = 0.0f;
  return z;
}

__device__ __forceinline__ v8f wmma_bf16(v16bf a, v16bf b, v8f c) {
  return __builtin_amdgcn_wmma_f32_16x16x32_bf16(
      /*neg_a=*/false, a, /*neg_b=*/false, b,
      /*c_mod=*/(short)0, c, /*reuse_a=*/false, /*reuse_b=*/false);
}

// A-fragment (16x32 bf16, row-major source, leading dim ld).
// Layout: lanes 0-15 rows 0-15 (K 0..15),
//      lanes 16-31 rows 0-15 (K 16..31). Pairs (k,k+1) contiguous.
__device__ __forceinline__ v16bf load_A_frag(const bf16_t* __restrict__ p, int ld) {
  const int lane = threadIdx.x & 31;
  const int row  = lane & 15;
  const int kb   = (lane & 16) ? 16 : 0;
  v16bf r;
#pragma unroll
  for (int i = 0; i < 8; ++i) {
    const int k = kb + 2 * i;
    r[2 * i]     = p[(size_t)row * ld + k];
    r[2 * i + 1] = p[(size_t)row * ld + k + 1];
  }
  return r;
}

// B-fragment (32x16 bf16) from row-major B[k][n], leading dim ld.
// Contiguous k order, as load_A_frag: lanes 0-15 (col = lane) hold K 0..15; lanes 16-31 hold K 16..31.
__device__ __forceinline__ v16bf load_B_frag(const bf16_t* __restrict__ p, int ld) {
  const int lane = threadIdx.x & 31;
  const int col  = lane & 15;
  const int kb   = (lane & 16);
  v16bf r;
#pragma unroll
  for (int i = 0; i < 8; ++i) {
    const int k = kb + 2 * i;
    r[2 * i]     = p[(size_t)k * ld + col];
    r[2 * i + 1] = p[(size_t)(k + 1) * ld + col];
  }
  return r;
}

// B-fragment where source is transposed: element (k,n) = p[n*ld + k].
// Pairs along k are contiguous -> b32 LDS gathers.
__device__ __forceinline__ v16bf load_Bt_frag(const bf16_t* __restrict__ p, int ld) {
  const int lane = threadIdx.x & 31;
  const int col  = lane & 15;
  const int kb   = (lane & 16);
  v16bf r;
#pragma unroll
  for (int i = 0; i < 8; ++i) {
    const int k = kb + 2 * i;
    r[2 * i]     = p[(size_t)col * ld + k];
    r[2 * i + 1] = p[(size_t)col * ld + k + 1];
  }
  return r;
}

// ---------------------------------------------------------------------------
// f32 -> bf16 convert (vectorized; all sizes are multiples of 1024)
// ---------------------------------------------------------------------------
__global__ __launch_bounds__(256) void f32_to_bf16_kernel(
    const float4* __restrict__ in, bf16_t* __restrict__ out, int n4) {
  int i = blockIdx.x * 256 + threadIdx.x;
  if (i < n4) {
    float4 v = in[i];
    out[4 * i + 0] = (bf16_t)v.x;
    out[4 * i + 1] = (bf16_t)v.y;
    out[4 * i + 2] = (bf16_t)v.z;
    out[4 * i + 3] = (bf16_t)v.w;
  }
}

// ---------------------------------------------------------------------------
// Tiled WMMA GEMM: C[M,N] = A[M,K] @ W[K,N] + bias  (A,W bf16; bias f32)
// Block tile 128x128x64, 256 threads = 8 waves (4x2 wave grid, 32x64 each).
// Double-buffered LDS; async global->LDS copies (CDNA5 ASYNCcnt path).
// ---------------------------------------------------------------------------
template <bool RELU, bool OUT_BF16>
__global__ __launch_bounds__(256) void gemm_bf16_kernel(
    const bf16_t* __restrict__ A, const bf16_t* __restrict__ W,
    const float* __restrict__ bias, void* __restrict__ Cout,
    int M, int N, int K) {
  __shared__ bf16_t As[2][128][72];   // 128x64 tile, +8 pad (144B row stride)
  __shared__ bf16_t Bs[2][64][136];   // 64x128 tile, +8 pad (272B row stride)

  const int tid  = threadIdx.x;
  const int lane = tid & 31;
  const int wave = tid >> 5;
  const int wm   = wave >> 1;           // 0..3 -> 32-row strip
  const int wn   = wave & 1;            // 0..1 -> 64-col strip
  const int rowBase = blockIdx.y * 128;
  const int colBase = blockIdx.x * 128;

  v8f acc[2][4];
#pragma unroll
  for (int s = 0; s < 2; ++s)
#pragma unroll
    for (int t = 0; t < 4; ++t) acc[s][t] = zero_v8f();

  const int arow = tid >> 1, acol = (tid & 1) * 32;  // 2 thr/row, 32 elems each
  const int brow = tid >> 2, bcol = (tid & 3) * 32;  // 4 thr/row, 32 elems each

  auto stage = [&](int buf, int kt) {
    const bf16_t* gA = A + (size_t)(rowBase + arow) * K + kt + acol;
    const bf16_t* gB = W + (size_t)(kt + brow) * N + colBase + bcol;
    // L2 prefetch two tiles ahead
    __builtin_prefetch(gA + 128, 0, 1);
    __builtin_prefetch(gB + (size_t)128 * N, 0, 1);
#if HAVE_ASYNC_COPY
#pragma unroll
    for (int j = 0; j < 4; ++j) {
      __builtin_amdgcn_global_load_async_to_lds_b128(
          (AS1 v4i*)(gA + j * 8), (AS3 v4i*)&As[buf][arow][acol + j * 8],
          0, 0);
      __builtin_amdgcn_global_load_async_to_lds_b128(
          (AS1 v4i*)(gB + j * 8), (AS3 v4i*)&Bs[buf][brow][bcol + j * 8],
          0, 0);
    }
#else
    const uint4* vA = reinterpret_cast<const uint4*>(gA);
    const uint4* vB = reinterpret_cast<const uint4*>(gB);
    uint4 ra[4], rb[4];
#pragma unroll
    for (int j = 0; j < 4; ++j) { ra[j] = vA[j]; rb[j] = vB[j]; }
#pragma unroll
    for (int j = 0; j < 4; ++j) {
      *reinterpret_cast<uint4*>(&As[buf][arow][acol + j * 8]) = ra[j];
      *reinterpret_cast<uint4*>(&Bs[buf][brow][bcol + j * 8]) = rb[j];
    }
#endif
  };

  const int ntiles = K >> 6;
  stage(0, 0);
  for (int kt = 0; kt < ntiles; ++kt) {
    const int cur = kt & 1;
    WAIT_ASYNC();
    __syncthreads();  // buffer `cur` visible to all; prior reads of `cur^1` done
    if (kt + 1 < ntiles) stage(cur ^ 1, (kt + 1) << 6);
#pragma unroll
    for (int kk = 0; kk < 2; ++kk) {
      v16bf af[2], bfrag[4];
#pragma unroll
      for (int s = 0; s < 2; ++s)
        af[s] = load_A_frag(&As[cur][wm * 32 + s * 16][kk * 32], 72);
#pragma unroll
      for (int t = 0; t < 4; ++t)
        bfrag[t] = load_B_frag(&Bs[cur][kk * 32][wn * 64 + t * 16], 136);
#pragma unroll
      for (int s = 0; s < 2; ++s)
#pragma unroll
        for (int t = 0; t < 4; ++t)
          acc[s][t] = wmma_bf16(af[s], bfrag[t], acc[s][t]);
    }
    // no tail barrier: `cur` is only overwritten by a stage() issued after the
    // NEXT iteration's top barrier, by which time all reads here are complete.
  }

  // Epilogue: C layout — col = lane&15; row = vgpr + (lane<16 ? 0 : 8)
  const int half8 = (lane & 16) ? 8 : 0;
  const int cN    = lane & 15;
#pragma unroll
  for (int s = 0; s < 2; ++s) {
#pragma unroll
    for (int t = 0; t < 4; ++t) {
      const int col = colBase + wn * 64 + t * 16 + cN;
      const float bv = bias ? bias[col] : 0.0f;
#pragma unroll
      for (int i = 0; i < 8; ++i) {
        const int row = rowBase + wm * 32 + s * 16 + i + half8;
        float v = acc[s][t][i] + bv;
        if (RELU) v = fmaxf(v, 0.0f);
        if (OUT_BF16)
          ((bf16_t*)Cout)[(size_t)row * N + col] = (bf16_t)v;
        else
          ((float*)Cout)[(size_t)row * N + col] = v;
      }
    }
  }
}

// ---------------------------------------------------------------------------
// Flash attention: softmax(Q K^T / 8) V per (batch, head).
// Grid (SEQ/64, BATCH*N_HEADS); 128 threads = 4 waves; wave owns 16 queries.
// K block staged row-major in LDS; V block staged TRANSPOSED so both WMMA
// operand gathers are contiguous-pair (b32) LDS reads.
// ---------------------------------------------------------------------------
__global__ __launch_bounds__(128) void attention_kernel(
    const bf16_t* __restrict__ Q, const bf16_t* __restrict__ Km,
    const bf16_t* __restrict__ V, bf16_t* __restrict__ ctx) {
  __shared__ bf16_t Ks[64][72];      // [key][d]  (9 KB)
  __shared__ bf16_t Vt[64][72];      // [d][key]  (9 KB, transposed V)
  __shared__ bf16_t Pt[4][16][72];   // per-wave P tiles (9 KB)

  const int tid  = threadIdx.x;
  const int lane = tid & 31;
  const int wave = tid >> 5;
  const int bh   = blockIdx.y;
  const int b    = bh >> 4;
  const int h    = bh & 15;
  const int q0   = blockIdx.x * 64 + wave * 16;

  const size_t qoff = (size_t)(b * SEQ + q0) * D_MODEL + h * HDIM;
  v16bf qf[2];
  qf[0] = load_A_frag(Q + qoff, D_MODEL);
  qf[1] = load_A_frag(Q + qoff + 32, D_MODEL);

  v8f o[4];
#pragma unroll
  for (int t = 0; t < 4; ++t) o[t] = zero_v8f();
  float m_i[8], l_i[8];
#pragma unroll
  for (int i = 0; i < 8; ++i) { m_i[i] = -INFINITY; l_i[i] = 0.0f; }

  const int half8 = (lane & 16) ? 8 : 0;
  const int cN    = lane & 15;
  const float scale = 0.125f;  // 1/sqrt(64)

  const int srow = tid >> 1;            // 0..63: staging row
  const int scol = (tid & 1) * 32;      // 2 threads per row, 32 elems each

  for (int kb = 0; kb < SEQ; kb += 64) {
    __syncthreads();  // previous block's reads of Ks/Vt complete
    // ---- stage K block (row-major) and V block (transposed) -----------
    {
      const uint4* gK = reinterpret_cast<const uint4*>(
          Km + (size_t)(b * SEQ + kb + srow) * D_MODEL + h * HDIM + scol);
      uint4 rk[4];
#pragma unroll
      for (int j = 0; j < 4; ++j) rk[j] = gK[j];
#pragma unroll
      for (int j = 0; j < 4; ++j)
        *reinterpret_cast<uint4*>(&Ks[srow][scol + j * 8]) = rk[j];

      const uint4* gV = reinterpret_cast<const uint4*>(
          V + (size_t)(b * SEQ + kb + srow) * D_MODEL + h * HDIM + scol);
      uint4 rv[4];
#pragma unroll
      for (int j = 0; j < 4; ++j) rv[j] = gV[j];
      const bf16_t* tv = reinterpret_cast<const bf16_t*>(rv);
#pragma unroll
      for (int j = 0; j < 32; ++j) Vt[scol + j][srow] = tv[j];  // transpose
    }
    __syncthreads();

    // ---- S = Q @ K^T (16 queries x 64 keys) ---------------------------
    v8f s[4];
#pragma unroll
    for (int t = 0; t < 4; ++t) s[t] = zero_v8f();
#pragma unroll
    for (int kk = 0; kk < 2; ++kk)
#pragma unroll
      for (int t = 0; t < 4; ++t) {
        v16bf bf = load_Bt_frag(&Ks[t * 16][kk * 32], 72);
        s[t] = wmma_bf16(qf[kk], bf, s[t]);
      }
#pragma unroll
    for (int t = 0; t < 4; ++t)
#pragma unroll
      for (int i = 0; i < 8; ++i) s[t][i] *= scale;

    // ---- online softmax ----------------------------------------------
    float p[4][8];
#pragma unroll
    for (int i = 0; i < 8; ++i) {
      float v = fmaxf(fmaxf(s[0][i], s[1][i]), fmaxf(s[2][i], s[3][i]));
#pragma unroll
      for (int m = 1; m <= 8; m <<= 1) v = fmaxf(v, __shfl_xor(v, m, 32));
      const float mn = fmaxf(m_i[i], v);
      const float alpha = __expf(m_i[i] - mn);
      m_i[i] = mn;
      float rs = 0.0f;
#pragma unroll
      for (int t = 0; t < 4; ++t) {
        const float e = __expf(s[t][i] - mn);
        p[t][i] = e;
        rs += e;
      }
#pragma unroll
      for (int m = 1; m <= 8; m <<= 1) rs += __shfl_xor(rs, m, 32);
      l_i[i] = l_i[i] * alpha + rs;
#pragma unroll
      for (int t = 0; t < 4; ++t) o[t][i] *= alpha;
    }

    // ---- stage P (bf16) into this wave's private LDS tile -------------
#pragma unroll
    for (int t = 0; t < 4; ++t)
#pragma unroll
      for (int i = 0; i < 8; ++i)
        Pt[wave][i + half8][t * 16 + cN] = (bf16_t)p[t][i];

    // ---- O += P @ V  (V gathered from transposed LDS tile) ------------
#pragma unroll
    for (int kk = 0; kk < 2; ++kk) {
      v16bf pa = load_A_frag(&Pt[wave][0][kk * 32], 72);
#pragma unroll
      for (int t = 0; t < 4; ++t) {
        v16bf bv = load_Bt_frag(&Vt[t * 16][kk * 32], 72);
        o[t] = wmma_bf16(pa, bv, o[t]);
      }
    }
  }

  // ---- normalize and write ctx (bf16, [token, D]) ----------------------
#pragma unroll
  for (int i = 0; i < 8; ++i) {
    const float inv = 1.0f / l_i[i];
    const int row = q0 + i + half8;
#pragma unroll
    for (int t = 0; t < 4; ++t)
      ctx[(size_t)(b * SEQ + row) * D_MODEL + h * HDIM + t * 16 + cN] =
          (bf16_t)(o[t][i] * inv);
  }
}

// ---------------------------------------------------------------------------
// Residual add + LayerNorm (unbiased variance, eps=1e-12). One block per row.
// ---------------------------------------------------------------------------
template <bool WRITE_BF16>
__global__ __launch_bounds__(256) void add_ln_kernel(
    const float* __restrict__ X, const float* __restrict__ Yadd,
    const float* __restrict__ gamma, const float* __restrict__ beta,
    float* __restrict__ outF, bf16_t* __restrict__ outB) {
  __shared__ float red[8];
  const int row = blockIdx.x;
  const int tid = threadIdx.x;
  const int lane = tid & 31, wave = tid >> 5;
  const float* xr = X + (size_t)row * D_MODEL;
  const float* yr = Yadd + (size_t)row * D_MODEL;

  float v[4];
#pragma unroll
  for (int j = 0; j < 4; ++j) v[j] = xr[tid + j * 256] + yr[tid + j * 256];

  float s = v[0] + v[1] + v[2] + v[3];
#pragma unroll
  for (int m = 1; m <= 16; m <<= 1) s += __shfl_xor(s, m, 32);
  if (lane == 0) red[wave] = s;
  __syncthreads();
  float tot = 0.0f;
#pragma unroll
  for (int w = 0; w < 8; ++w) tot += red[w];
  const float mean = tot * (1.0f / D_MODEL);
  __syncthreads();

  float ss = 0.0f;
#pragma unroll
  for (int j = 0; j < 4; ++j) {
    const float d = v[j] - mean;
    ss += d * d;
  }
#pragma unroll
  for (int m = 1; m <= 16; m <<= 1) ss += __shfl_xor(ss, m, 32);
  if (lane == 0) red[wave] = ss;
  __syncthreads();
  float tss = 0.0f;
#pragma unroll
  for (int w = 0; w < 8; ++w) tss += red[w];
  const float var = tss * (1.0f / (D_MODEL - 1));  // ddof=1 (torch unbiased)
  const float inv = rsqrtf(var + LN_EPS);

#pragma unroll
  for (int j = 0; j < 4; ++j) {
    const int c = tid + j * 256;
    const float r = (v[j] - mean) * inv * gamma[c] + beta[c];
    outF[(size_t)row * D_MODEL + c] = r;
    if (WRITE_BF16) outB[(size_t)row * D_MODEL + c] = (bf16_t)r;
  }
}

// ---------------------------------------------------------------------------
// Orchestration. Workspace map (~152 MB).
// ---------------------------------------------------------------------------
#define MB (size_t)(1u << 20)

extern "C" void kernel_launch(void* const* d_in, const int* in_sizes, int n_in,
                              void* d_out, int out_size, void* d_ws, size_t ws_size,
                              hipStream_t stream) {
  const float* x      = (const float*)d_in[0];
  const float* wq     = (const float*)d_in[1];
  const float* bq     = (const float*)d_in[2];
  const float* wk     = (const float*)d_in[3];
  const float* bk     = (const float*)d_in[4];
  const float* wv     = (const float*)d_in[5];
  const float* bv     = (const float*)d_in[6];
  const float* wo     = (const float*)d_in[7];
  const float* bo     = (const float*)d_in[8];
  const float* gamma1 = (const float*)d_in[9];
  const float* beta1  = (const float*)d_in[10];
  const float* w1     = (const float*)d_in[11];
  const float* b1     = (const float*)d_in[12];
  const float* w2     = (const float*)d_in[13];
  const float* b2     = (const float*)d_in[14];
  const float* gamma2 = (const float*)d_in[15];
  const float* beta2  = (const float*)d_in[16];

  char* ws = (char*)d_ws;
  bf16_t* xb   = (bf16_t*)(ws + 0);
  bf16_t* wqb  = (bf16_t*)(ws + 8 * MB);
  bf16_t* wkb  = (bf16_t*)(ws + 10 * MB);
  bf16_t* wvb  = (bf16_t*)(ws + 12 * MB);
  bf16_t* wob  = (bf16_t*)(ws + 14 * MB);
  bf16_t* w1b  = (bf16_t*)(ws + 16 * MB);
  bf16_t* w2b  = (bf16_t*)(ws + 24 * MB);
  bf16_t* qb   = (bf16_t*)(ws + 32 * MB);
  bf16_t* kb   = (bf16_t*)(ws + 40 * MB);
  bf16_t* vb   = (bf16_t*)(ws + 48 * MB);
  bf16_t* ctxb = (bf16_t*)(ws + 56 * MB);
  float*  attn = (float*)(ws + 64 * MB);
  float*  x1f  = (float*)(ws + 80 * MB);
  bf16_t* x1b  = (bf16_t*)(ws + 96 * MB);
  bf16_t* hid  = (bf16_t*)(ws + 104 * MB);
  float*  ffn  = (float*)(ws + 136 * MB);

  auto conv = [&](const float* src, bf16_t* dst, int n) {
    int n4 = n / 4;
    f32_to_bf16_kernel<<<(n4 + 255) / 256, 256, 0, stream>>>(
        (const float4*)src, dst, n4);
  };

  conv(x, xb, NTOK * D_MODEL);
  conv(wq, wqb, D_MODEL * D_MODEL);
  conv(wk, wkb, D_MODEL * D_MODEL);
  conv(wv, wvb, D_MODEL * D_MODEL);
  conv(wo, wob, D_MODEL * D_MODEL);
  conv(w1, w1b, D_MODEL * HIDDEN);
  conv(w2, w2b, HIDDEN * D_MODEL);

  const dim3 blk(256);
  {
    dim3 grid(D_MODEL / 128, NTOK / 128);
    gemm_bf16_kernel<false, true><<<grid, blk, 0, stream>>>(xb, wqb, bq, qb, NTOK, D_MODEL, D_MODEL);
    gemm_bf16_kernel<false, true><<<grid, blk, 0, stream>>>(xb, wkb, bk, kb, NTOK, D_MODEL, D_MODEL);
    gemm_bf16_kernel<false, true><<<grid, blk, 0, stream>>>(xb, wvb, bv, vb, NTOK, D_MODEL, D_MODEL);
  }
  {
    dim3 grid(SEQ / 64, BATCH * N_HEADS);
    attention_kernel<<<grid, dim3(128), 0, stream>>>(qb, kb, vb, ctxb);
  }
  {
    dim3 grid(D_MODEL / 128, NTOK / 128);
    gemm_bf16_kernel<false, false><<<grid, blk, 0, stream>>>(ctxb, wob, bo, attn, NTOK, D_MODEL, D_MODEL);
  }
  add_ln_kernel<true><<<NTOK, blk, 0, stream>>>(x, attn, gamma1, beta1, x1f, x1b);
  {
    dim3 grid(HIDDEN / 128, NTOK / 128);
    gemm_bf16_kernel<true, true><<<grid, blk, 0, stream>>>(x1b, w1b, b1, hid, NTOK, HIDDEN, D_MODEL);
  }
  {
    dim3 grid(D_MODEL / 128, NTOK / 128);
    gemm_bf16_kernel<false, false><<<grid, blk, 0, stream>>>(hid, w2b, b2, ffn, NTOK, D_MODEL, HIDDEN);
  }
  add_ln_kernel<false><<<NTOK, blk, 0, stream>>>(x1f, ffn, gamma2, beta2, (float*)d_out, nullptr);
}
